// LSTM_78039555768548
// MI455X (gfx1250) — matrix operand order repaired and checked by reading, not compiled, not run on hardware
//
#include <hip/hip_runtime.h>
#include <stdint.h>

// ---------------------------------------------------------------------------
// Types matching CDNA5 WMMA operand shapes (wave32)
// ---------------------------------------------------------------------------
typedef __attribute__((ext_vector_type(16))) __bf16   v16bf;  // A/B fragment: 16 bf16 = 8 VGPRs
typedef __attribute__((ext_vector_type(8)))  float    v8f;    // C/D fragment: 8 f32  = 8 VGPRs
typedef __attribute__((ext_vector_type(4)))  unsigned v4u;    // 16-byte load unit
typedef int v4i_vec __attribute__((vector_size(16)));         // type expected by async-LDS builtin

union Frag16 {
  v4u   u[2];
  v16bf bf;
};

#define B_   64
#define L_   512
#define DIN  1024
#define D_   1024
#define TD   3072          // 3*D
#define NWG_REC 32         // persistent workgroups for the recurrence
#define COLS_PER_WG 32     // h-columns owned per workgroup (3*32=96 gate cols)
#define SWR_STRIDE 1032    // padded K-stride (halfwords) for Wr slice in LDS
#define BAR_STRIDE 16      // dwords between per-WG flags (separate 64B lines)

// ---------------------------------------------------------------------------
// Helpers
// ---------------------------------------------------------------------------
__device__ __forceinline__ uint16_t f32_to_bf16(float f) {
  uint32_t u = __builtin_bit_cast(uint32_t, f);
  uint32_t r = ((u >> 16) & 1u) + 0x7FFFu;   // round-to-nearest-even
  return (uint16_t)((u + r) >> 16);
}

// 3-op float pair -> packed bf16x2 (round-half-up): 2x v_add + v_perm_b32
__device__ __forceinline__ uint32_t pack_bf16x2_fast(float lo, float hi) {
  uint32_t ul = __builtin_bit_cast(uint32_t, lo) + 0x8000u;
  uint32_t uh = __builtin_bit_cast(uint32_t, hi) + 0x8000u;
  return __builtin_amdgcn_perm(uh, ul, 0x07060302u);
}

__device__ __forceinline__ float fast_sigmoid(float x) {
  // sigmoid(x) = 1 / (1 + 2^(-x*log2(e)))  -> v_exp_f32 + v_rcp_f32
  return __builtin_amdgcn_rcpf(1.0f + __builtin_amdgcn_exp2f(-1.44269504088896f * x));
}
__device__ __forceinline__ float fast_tanh(float x) {
  return 2.0f * fast_sigmoid(2.0f * x) - 1.0f;
}

// ---------------------------------------------------------------------------
// Tiled transpose + fp32->bf16: out[col][k] = bf16(in[k][3072-wide col])
// Both global sides fully coalesced; 33-wide LDS tile avoids bank conflicts.
// ---------------------------------------------------------------------------
__global__ __launch_bounds__(256) void k_cvt_transpose(const float* __restrict__ in,
                                                       uint16_t* __restrict__ out) {
  __shared__ uint16_t tile[32][33];
  const int bc = blockIdx.x * 32;          // column base (fast dim of input)
  const int bk = blockIdx.y * 32;          // k base (row of input)
  const int tx = threadIdx.x & 31;
  const int ty = threadIdx.x >> 5;         // 0..7
  for (int i = 0; i < 4; ++i) {
    const int k = bk + ty + i * 8;
    tile[ty + i * 8][tx] = f32_to_bf16(in[(size_t)k * TD + bc + tx]);
  }
  __syncthreads();
  for (int i = 0; i < 4; ++i) {
    const int c = bc + ty + i * 8;
    out[(size_t)c * DIN + bk + tx] = tile[tx][ty + i * 8];
  }
}

// ---------------------------------------------------------------------------
// init: zero per-step/per-WG barrier flags, seed h_buf slot0 with bf16(h0)
// (re-run every launch so graph replays start from a clean barrier state)
// ---------------------------------------------------------------------------
__global__ __launch_bounds__(256) void k_init(const float* __restrict__ h0,
                                              uint16_t* __restrict__ hbuf,
                                              unsigned* __restrict__ bar) {
  const int stride = gridDim.x * 256;
  const int i = blockIdx.x * 256 + threadIdx.x;
  for (int j = i; j < B_ * D_; j += stride) hbuf[j] = f32_to_bf16(h0[j]);
  for (int j = i; j < L_ * NWG_REC * BAR_STRIDE; j += stride) bar[j] = 0u;
}

// ---------------------------------------------------------------------------
// x_proj[BL,3072] = bf16(x)[BL,1024] @ Wk[1024,3072]  (fp32 accumulate)
// Wk pre-transposed to [col][k] bf16 -> B fragments load straight from global
// (two b128 per fragment, L2-resident). A built from float4 loads + v_perm
// conversion. No LDS, no barriers: pure global -> regs -> v_wmma.
// Workgroup tile 128x128, 8 waves, each wave 2 M-tiles x 4 N-tiles.
// ---------------------------------------------------------------------------
__global__ __launch_bounds__(256) void k_gemm_xproj(const float* __restrict__ x,
                                                    const uint16_t* __restrict__ wkT,
                                                    float* __restrict__ xp) {
  const int tid  = threadIdx.x;
  const int lane = tid & 31;
  const int w    = tid >> 5;
  const int col0 = blockIdx.x * 128;
  const int row0 = blockIdx.y * 128;

  const int mt0  = 2 * (w & 3);
  const int nt0  = 4 * (w >> 2);
  const int half = lane >> 4;
  const int ln   = lane & 15;

  // per-lane base pointers
  const float* arow[2];
  for (int mi = 0; mi < 2; ++mi)
    arow[mi] = x + (size_t)(row0 + (mt0 + mi) * 16 + ln) * DIN + half * 8;
  const uint16_t* brow[4];
  for (int ni = 0; ni < 4; ++ni)
    brow[ni] = wkT + (size_t)(col0 + (nt0 + ni) * 16 + ln) * DIN + half * 8;

  v8f acc[2][4];
  const v8f vzero = {0.f, 0.f, 0.f, 0.f, 0.f, 0.f, 0.f, 0.f};
  for (int mi = 0; mi < 2; ++mi)
    for (int ni = 0; ni < 4; ++ni) acc[mi][ni] = vzero;

  for (int kb = 0; kb < DIN; kb += 32) {
    Frag16 a[2], b[4];
    for (int ni = 0; ni < 4; ++ni) {
      const v4u* p = (const v4u*)(brow[ni] + kb);   // K {0..7 | 8..15}, {16..23 | 24..31}
      b[ni].u[0] = p[0];
      b[ni].u[1] = p[2];
    }
    for (int mi = 0; mi < 2; ++mi) {
      const float* pa = arow[mi] + kb;
      const float4 f0 = *(const float4*)(pa);        // K {0..3 | 8..11}
      const float4 f1 = *(const float4*)(pa + 4);    // K {4..7 | 12..15}
      const float4 f2 = *(const float4*)(pa + 16);   // K {16..19 | 24..27}
      const float4 f3 = *(const float4*)(pa + 20);   // K {20..23 | 28..31}
      v4u u0, u1;
      u0.x = pack_bf16x2_fast(f0.x, f0.y);
      u0.y = pack_bf16x2_fast(f0.z, f0.w);
      u0.z = pack_bf16x2_fast(f1.x, f1.y);
      u0.w = pack_bf16x2_fast(f1.z, f1.w);
      u1.x = pack_bf16x2_fast(f2.x, f2.y);
      u1.y = pack_bf16x2_fast(f2.z, f2.w);
      u1.z = pack_bf16x2_fast(f3.x, f3.y);
      u1.w = pack_bf16x2_fast(f3.z, f3.w);
      a[mi].u[0] = u0;
      a[mi].u[1] = u1;
    }
    for (int mi = 0; mi < 2; ++mi)
      for (int ni = 0; ni < 4; ++ni)
        acc[mi][ni] = __builtin_amdgcn_wmma_f32_16x16x32_bf16(
            false, a[mi].bf, false, b[ni].bf, (short)0, acc[mi][ni], false, false);
  }

  // epilogue: per VGPR r, lanes 0-15 / 16-31 each write a contiguous 64B run
  for (int mi = 0; mi < 2; ++mi)
    for (int ni = 0; ni < 4; ++ni) {
      const int colg  = col0 + (nt0 + ni) * 16 + ln;
      const int rbase = row0 + (mt0 + mi) * 16 + half * 8;
      for (int r = 0; r < 8; ++r)
        xp[(size_t)(rbase + r) * TD + colg] = acc[mi][ni][r];
    }
}

// ---------------------------------------------------------------------------
// Persistent recurrence: 32 WGs, each owns 32 h-columns (96 gate cols).
// Wr slice (~194KB, [col][k] padded) lives in LDS for all 512 steps; c stays
// in registers as a WMMA C-fragment. Per step: gates = x_proj[:,t] + h @ Wr.
// Device-wide step sync: per-WG release-flag + per-lane spin (wave 0), with
// xp prefetch and out-stores moved into the sync shadow.
// ---------------------------------------------------------------------------
__global__ __launch_bounds__(256) void k_lstm_rec(const float* __restrict__ xp,
                                                  const uint16_t* __restrict__ wrT,
                                                  const float* __restrict__ c0,
                                                  uint16_t* __restrict__ hbuf,
                                                  unsigned* __restrict__ bar,
                                                  float* __restrict__ out) {
  extern __shared__ uint16_t sWr[];  // [96][SWR_STRIDE] bf16 (conflict-free b128 reads)

  const int tid  = threadIdx.x;
  const int lane = tid & 31;
  const int w    = tid >> 5;
  const int g    = blockIdx.x;
  const int d0   = g * COLS_PER_WG;

  // ---- one-time fill of this WG's Wr columns (wrT is [3072 cols][1024 k]):
  // contiguous 16B chunks on both sides -> coalesced / conflict-free.
  // Uses the CDNA5 async global->LDS path when the builtin is available.
  for (int c = tid; c < 96 * 128; c += 256) {
    const int p    = c >> 7;          // packed column 0..95
    const int off  = (c & 127) * 8;   // halfword offset within column
    const int gate = p >> 5;
    const int coff = p & 31;
    const uint16_t* gsrc = wrT + (size_t)(gate * D_ + d0 + coff) * DIN + off;
    uint16_t*       ldst = sWr + p * SWR_STRIDE + off;
#if __has_builtin(__builtin_amdgcn_global_load_async_to_lds_b128)
    __builtin_amdgcn_global_load_async_to_lds_b128(
        (v4i_vec*)gsrc, (v4i_vec*)ldst, 0, 0);
#else
    *(v4u*)ldst = *(const v4u*)gsrc;
#endif
  }
#if __has_builtin(__builtin_amdgcn_global_load_async_to_lds_b128)
  asm volatile("s_wait_asynccnt 0" ::: "memory");
#endif
  __syncthreads();

  const int mt    = w & 3;                     // batch 16-block
  const int nt    = w >> 2;                    // column 16-block (0..1)
  const int half  = lane >> 4;
  const int ln    = lane & 15;
  const int mbase = mt * 16 + half * 8;        // C-fragment row base (batch)
  const int colh  = d0 + nt * 16 + ln;         // global h column
  const int cf    = colh;                      // f gate column in [0,3072)
  const int co    = D_ + colh;
  const int cg    = 2 * D_ + colh;
  const int pf    = nt * 16 + ln;              // packed LDS column indices
  const int po    = 32 + pf;
  const int pg    = 64 + pf;
  const int arow  = mt * 16 + ln;              // A-fragment row (batch)

  // ---- c lives in registers as a C-fragment for all 512 steps
  v8f cfrag;
  for (int r = 0; r < 8; ++r) cfrag[r] = c0[(size_t)(mbase + r) * D_ + colh];

  auto preload = [&](int tt, v8f& aF, v8f& aO, v8f& aG) {
    const size_t xt = (size_t)tt * TD;
    for (int r = 0; r < 8; ++r) {
      const size_t rowoff = (size_t)(mbase + r) * L_ * TD + xt;
      aF[r] = xp[rowoff + cf];
      aO[r] = xp[rowoff + co];
      aG[r] = xp[rowoff + cg];
    }
  };

  v8f accF, accO, accG;
  preload(0, accF, accO, accG);   // gates = xp + h@Wr

  for (int t = 0; t < L_; ++t) {
    if (t > 0) {
      if (w == 0) {  // wave 0: each lane polls one WG's flag (parallel L2 reads)
        const unsigned* f = bar + (size_t)(t - 1) * (NWG_REC * BAR_STRIDE) + lane * BAR_STRIDE;
        while (__hip_atomic_load(f, __ATOMIC_RELAXED, __HIP_MEMORY_SCOPE_AGENT) == 0u)
          __builtin_amdgcn_s_sleep(1);
      }
      __syncthreads();
      __builtin_amdgcn_fence(__ATOMIC_ACQUIRE, "agent");
    }
    const uint16_t* hsrc = hbuf + (size_t)(t & 1) * B_ * D_;
    uint16_t*       hdst = hbuf + (size_t)((t + 1) & 1) * B_ * D_;

    // K-loop: A(h) fragments straight from global (L2-resident), B from LDS
    for (int kb = 0; kb < D_; kb += 32) {
      Frag16 a, bF, bO, bG;
      const v4u* pa = (const v4u*)(hsrc + (size_t)arow * D_ + kb + half * 16);
      a.u[0] = pa[0];
      a.u[1] = pa[1];
      const v4u* pbf = (const v4u*)(sWr + pf * SWR_STRIDE + kb + half * 16);
      bF.u[0] = pbf[0]; bF.u[1] = pbf[1];
      const v4u* pbo = (const v4u*)(sWr + po * SWR_STRIDE + kb + half * 16);
      bO.u[0] = pbo[0]; bO.u[1] = pbo[1];
      const v4u* pbg = (const v4u*)(sWr + pg * SWR_STRIDE + kb + half * 16);
      bG.u[0] = pbg[0]; bG.u[1] = pbg[1];
      accF = __builtin_amdgcn_wmma_f32_16x16x32_bf16(false, a.bf, false, bF.bf,
                                                     (short)0, accF, false, false);
      accO = __builtin_amdgcn_wmma_f32_16x16x32_bf16(false, a.bf, false, bO.bf,
                                                     (short)0, accO, false, false);
      accG = __builtin_amdgcn_wmma_f32_16x16x32_bf16(false, a.bf, false, bG.bf,
                                                     (short)0, accG, false, false);
    }

    // gate nonlinearity + state update in registers; publish only h (bf16)
    v8f cnew, hvals;
    for (int r = 0; r < 8; ++r) {
      const float sf = fast_sigmoid(accF[r]);
      const float cn = cfrag[r] * sf + (1.0f - sf) * fast_tanh(accG[r]);
      const float hn = fast_sigmoid(accO[r]) * fast_tanh(cn);
      cnew[r] = cn;
      hvals[r] = hn;
      hdst[(size_t)(mbase + r) * D_ + colh] = f32_to_bf16(hn);
    }

    // release h(t+1), then signal this WG's flag for step t
    __builtin_amdgcn_fence(__ATOMIC_RELEASE, "agent");
    __syncthreads();
    if (tid == 0)
      __hip_atomic_store(bar + (size_t)t * (NWG_REC * BAR_STRIDE) + g * BAR_STRIDE,
                         1u, __ATOMIC_RELEASE, __HIP_MEMORY_SCOPE_AGENT);

    // work nobody else waits on, done in the sync shadow:
    for (int r = 0; r < 8; ++r)
      out[((size_t)(mbase + r) * L_ + t) * D_ + colh] = hvals[r];
    cfrag = cnew;
    if (t + 1 < L_) preload(t + 1, accF, accO, accG);
  }
}

// ---------------------------------------------------------------------------
// launch
// ---------------------------------------------------------------------------
extern "C" void kernel_launch(void* const* d_in, const int* in_sizes, int n_in,
                              void* d_out, int out_size, void* d_ws, size_t ws_size,
                              hipStream_t stream) {
  (void)in_sizes; (void)n_in; (void)out_size; (void)ws_size;
  const float* x  = (const float*)d_in[0];
  const float* Wk = (const float*)d_in[1];
  const float* Wr = (const float*)d_in[2];
  const float* c0 = (const float*)d_in[3];
  const float* h0 = (const float*)d_in[4];
  float* out = (float*)d_out;

  // workspace layout
  char* ws = (char*)d_ws;
  const size_t WT_BYTES   = (size_t)DIN * TD * 2;                  // 6 MB each
  const size_t HBUF_BYTES = (size_t)2 * B_ * D_ * 2;               // 256 KB
  const size_t BAR_BYTES  = (size_t)L_ * NWG_REC * BAR_STRIDE * 4; // 1 MB
  uint16_t* wkT  = (uint16_t*)ws;
  uint16_t* wrT  = (uint16_t*)(ws + WT_BYTES);
  uint16_t* hbuf = (uint16_t*)(ws + 2 * WT_BYTES);
  unsigned* bar  = (unsigned*)(ws + 2 * WT_BYTES + HBUF_BYTES);
  float*    xp   = (float*)(ws + 2 * WT_BYTES + HBUF_BYTES + BAR_BYTES);

  dim3 tgrid(TD / 32, DIN / 32);  // 96 x 32
  k_cvt_transpose<<<tgrid, 256, 0, stream>>>(Wk, wkT);
  k_cvt_transpose<<<tgrid, 256, 0, stream>>>(Wr, wrT);
  k_init<<<256, 256, 0, stream>>>(h0, hbuf, bar);

  dim3 ggrid(TD / 128, (B_ * L_) / 128);   // 24 x 256 workgroups
  k_gemm_xproj<<<ggrid, 256, 0, stream>>>(x, wkT, xp);

  k_lstm_rec<<<NWG_REC, 256, 96 * SWR_STRIDE * sizeof(uint16_t), stream>>>(
      xp, wrT, c0, hbuf, bar, out);
}
